// HybridMambaNetwork_23132693856728
// MI455X (gfx1250) — matrix operand order repaired and checked by reading, not compiled, not run on hardware
//
#include <hip/hip_runtime.h>
#include <hip/hip_bf16.h>

// Problem constants (from reference)
#define B_   128
#define NA_  8
#define AD_  16
#define DM_  512
#define DI_  1024
#define DS_  32
#define DC_  4
#define DR_  32
#define NB_  4
#define OD_  128
#define APK_ 32    // a_prev padded K (real AD_=16, zero padded to 32)
#define DBCP_ 128  // dbc row stride padded 96 -> 128 (N multiple of 64)
#define HDP_  64   // head N padded 16 -> 64

typedef __attribute__((ext_vector_type(16))) _Float16 v16h;
typedef __attribute__((ext_vector_type(8)))  _Float16 v8h;
typedef __attribute__((ext_vector_type(8)))  float    v8f;

union V16U { v16h v; v8h h8[2]; };

__device__ __forceinline__ float sigm_(float x) { return 1.f / (1.f + __expf(-x)); }
__device__ __forceinline__ float softplus_(float x) { return (x > 20.f) ? x : log1pf(__expf(x)); }

// ---------------------------------------------------------------------------
// WMMA GEMM: C[M,N](f32) = (beta?C:0) + A[M,K](f16,row-major) @ B[K,N]
// B supplied TRANSPOSED as Bt[N][K] (f16): every lane's operand fragment is
// two contiguous 16-byte loads (global_load_b128). One wave = 16x64 tile
// (4 x 16x16 WMMA, A fragment reused 4x). N MUST be a multiple of 64 and K a
// multiple of 32 (callers pad) -> zero per-lane guards, EXEC all-ones at
// every WMMA, unconditional straight-line k-loop.
// Lane layout (wave32; A/B use this code's contiguous k order, C as in CDNA5 ISA 7.12.2):
//   A 16x32 : m = lane&15 ; halves [j]    = k0+16*(lane>>4)+j  (contig)
//   B 32x16 : n = lane&15 ; halves [j]    = k0+16*(lane>>4)+j  (contig in Bt)
//   C 16x16 : n = lane&15 ; vgpr v -> m = v + 8*(lane>>4)
// ---------------------------------------------------------------------------
__global__ void wmma_gemm_kernel(const _Float16* __restrict__ A, int lda,
                                 const _Float16* __restrict__ Bt, int ldbt,
                                 float* __restrict__ C, int ldc,
                                 const float* __restrict__ bias,
                                 int M, int N64, int K, int beta, int act)
{
    const int lane   = threadIdx.x & 31;
    const int tilesN = N64 >> 6;
    const int wave   = blockIdx.x * (blockDim.x >> 5) + (threadIdx.x >> 5);
    if (wave >= (M >> 4) * tilesN) return;     // wave-uniform exit

    const int tm  = (wave / tilesN) << 4;
    const int tn0 = (wave % tilesN) << 6;
    const int h   = lane >> 4;                 // 0 or 1
    const int l16 = lane & 15;

    const _Float16* Ap  = A  + (size_t)(tm + l16) * lda + 16 * h;
    const _Float16* Bp0 = Bt + (size_t)(tn0 +  0 + l16) * ldbt + 16 * h;
    const _Float16* Bp1 = Bt + (size_t)(tn0 + 16 + l16) * ldbt + 16 * h;
    const _Float16* Bp2 = Bt + (size_t)(tn0 + 32 + l16) * ldbt + 16 * h;
    const _Float16* Bp3 = Bt + (size_t)(tn0 + 48 + l16) * ldbt + 16 * h;

    v8f acc0 = {}, acc1 = {}, acc2 = {}, acc3 = {};
    if (beta) {
#pragma unroll
        for (int v = 0; v < 8; ++v) {
            size_t row = (size_t)(tm + v + 8 * h) * ldc + tn0 + l16;
            acc0[v] = C[row];
            acc1[v] = C[row + 16];
            acc2[v] = C[row + 32];
            acc3[v] = C[row + 48];
        }
    }

    for (int k0 = 0; k0 < K; k0 += 32) {
        __builtin_prefetch(Ap + k0 + 64, 0, 3);          // global_prefetch_b8 (near)
        V16U a, b0, b1, b2, b3;
        a.h8[0]  = *(const v8h*)(Ap  + k0);
        a.h8[1]  = *(const v8h*)(Ap  + k0 + 8);
        b0.h8[0] = *(const v8h*)(Bp0 + k0);
        b0.h8[1] = *(const v8h*)(Bp0 + k0 + 8);
        b1.h8[0] = *(const v8h*)(Bp1 + k0);
        b1.h8[1] = *(const v8h*)(Bp1 + k0 + 8);
        b2.h8[0] = *(const v8h*)(Bp2 + k0);
        b2.h8[1] = *(const v8h*)(Bp2 + k0 + 8);
        b3.h8[0] = *(const v8h*)(Bp3 + k0);
        b3.h8[1] = *(const v8h*)(Bp3 + k0 + 8);
        acc0 = __builtin_amdgcn_wmma_f32_16x16x32_f16(false, a.v, false, b0.v,
                                                      (short)0, acc0, false, false);
        acc1 = __builtin_amdgcn_wmma_f32_16x16x32_f16(false, a.v, false, b1.v,
                                                      (short)0, acc1, false, false);
        acc2 = __builtin_amdgcn_wmma_f32_16x16x32_f16(false, a.v, false, b2.v,
                                                      (short)0, acc2, false, false);
        acc3 = __builtin_amdgcn_wmma_f32_16x16x32_f16(false, a.v, false, b3.v,
                                                      (short)0, acc3, false, false);
    }

#pragma unroll
    for (int j = 0; j < 4; ++j) {
        const v8f* accp = (j == 0) ? &acc0 : (j == 1) ? &acc1 : (j == 2) ? &acc2 : &acc3;
        int   n  = tn0 + 16 * j + l16;
        float bz = bias ? bias[n] : 0.f;
#pragma unroll
        for (int v = 0; v < 8; ++v) {
            float r = (*accp)[v] + bz;
            if (act == 1) r = softplus_(r);
            C[(size_t)(tm + v + 8 * h) * ldc + n] = r;
        }
    }
}

// ---------------------------------------------------------------------------
// LayerNorm over DM=512 with scale, optional ctx add AFTER scaling; f16 out.
// One block (256 threads) per batch row; LDS tree reduction.
// ---------------------------------------------------------------------------
__global__ void ln_kernel(const float* __restrict__ x, const float* __restrict__ scale,
                          const float* __restrict__ ctx, _Float16* __restrict__ out)
{
    __shared__ float ssum[256];
    __shared__ float ssq[256];
    const int b = blockIdx.x;
    const int t = threadIdx.x;
    float v0 = x[(size_t)b * DM_ + t];
    float v1 = x[(size_t)b * DM_ + 256 + t];
    ssum[t] = v0 + v1;
    ssq[t]  = v0 * v0 + v1 * v1;
    __syncthreads();
    for (int off = 128; off > 0; off >>= 1) {
        if (t < off) { ssum[t] += ssum[t + off]; ssq[t] += ssq[t + off]; }
        __syncthreads();
    }
    float mean = ssum[0] * (1.f / DM_);
    float var  = ssq[0] * (1.f / DM_) - mean * mean;
    float inv  = rsqrtf(var + 1e-5f);
    float o0 = (v0 - mean) * inv * scale[t];
    float o1 = (v1 - mean) * inv * scale[256 + t];
    if (ctx) {
        o0 += ctx[(size_t)b * DM_ + t];
        o1 += ctx[(size_t)b * DM_ + 256 + t];
    }
    out[(size_t)b * DM_ + t]       = (_Float16)o0;
    out[(size_t)b * DM_ + 256 + t] = (_Float16)o1;
}

// ---------------------------------------------------------------------------
// Depthwise conv buffer shift + SiLU:  xc = silu(sum_k buf'[k,d]*cw[k,d]+cb[d])
// ---------------------------------------------------------------------------
__global__ void conv_kernel(float* __restrict__ buf, const float* __restrict__ xz,
                            const float* __restrict__ cw, const float* __restrict__ cb,
                            float* __restrict__ xcf, _Float16* __restrict__ xch)
{
    int idx = blockIdx.x * blockDim.x + threadIdx.x;
    if (idx >= B_ * DI_) return;
    int b = idx >> 10, d = idx & (DI_ - 1);
    float* bp = buf + ((size_t)b * DC_) * DI_ + d;
    float b1 = bp[1 * DI_], b2 = bp[2 * DI_], b3 = bp[3 * DI_];
    float xn = xz[(size_t)b * (2 * DI_) + d];
    bp[0 * DI_] = b1; bp[1 * DI_] = b2; bp[2 * DI_] = b3; bp[3 * DI_] = xn;
    float s = b1 * cw[d] + b2 * cw[DI_ + d] + b3 * cw[2 * DI_ + d] + xn * cw[3 * DI_ + d] + cb[d];
    float r = s * sigm_(s);
    xcf[idx] = r;
    xch[idx] = (_Float16)r;
}

// ---------------------------------------------------------------------------
// SSM state update + C-readout + z-gating. One thread per (b,d); DS=32 float4.
// dbc rows are DBCP_(=128)-strided: dt @ col0, Bm @ col32, Cm @ col64.
// ---------------------------------------------------------------------------
__global__ void state_kernel(float* __restrict__ h, const float* __restrict__ A_log,
                             const float* __restrict__ Dw, const float* __restrict__ dbc,
                             const float* __restrict__ delta, const float* __restrict__ xcf,
                             const float* __restrict__ xz, _Float16* __restrict__ ygh)
{
    int idx = blockIdx.x * blockDim.x + threadIdx.x;
    if (idx >= B_ * DI_) return;
    int b = idx >> 10, d = idx & (DI_ - 1);
    float del = delta[idx];
    float xcv = xcf[idx];
    float du  = del * xcv;
    const float* Bm = dbc + (size_t)b * DBCP_ + DR_;
    const float* Cm = dbc + (size_t)b * DBCP_ + DR_ + DS_;
    float* hp = h + (size_t)idx * DS_;
    const float* ap = A_log + (size_t)d * DS_;
    float y = 0.f;
#pragma unroll
    for (int s = 0; s < DS_; s += 4) {
        float4 hv = *(float4*)(hp + s);
        float4 al = *(const float4*)(ap + s);
        float dA0 = __expf(-__expf(al.x) * del);
        float dA1 = __expf(-__expf(al.y) * del);
        float dA2 = __expf(-__expf(al.z) * del);
        float dA3 = __expf(-__expf(al.w) * del);
        hv.x = dA0 * hv.x + du * Bm[s + 0];
        hv.y = dA1 * hv.y + du * Bm[s + 1];
        hv.z = dA2 * hv.z + du * Bm[s + 2];
        hv.w = dA3 * hv.w + du * Bm[s + 3];
        y += hv.x * Cm[s + 0] + hv.y * Cm[s + 1] + hv.z * Cm[s + 2] + hv.w * Cm[s + 3];
        *(float4*)(hp + s) = hv;
    }
    y += Dw[d] * xcv;
    float z = xz[(size_t)b * (2 * DI_) + DI_ + d];
    ygh[idx] = (_Float16)(y * (z * sigm_(z)));
}

// ---------------------------------------------------------------------------
// Head: raw/logprob/tanh; writes outputs for agent i and padded a_prev (f16).
// mean rows are HDP_(=64)-strided.
// ---------------------------------------------------------------------------
__global__ void head_kernel(const float* __restrict__ mean, const float* __restrict__ eps,
                            const float* __restrict__ log_std,
                            float* __restrict__ acts, float* __restrict__ logs,
                            float* __restrict__ raws, _Float16* __restrict__ aph, int agent)
{
    int b = blockIdx.x * blockDim.x + threadIdx.x;
    if (b >= B_) return;
    const float log2pi = 1.8378770664093453f;
    const float ln2    = 0.6931471805599453f;
    float lp = -0.5f * (float)AD_ * log2pi;
#pragma unroll
    for (int j = 0; j < AD_; ++j) {
        float st  = softplus_(log_std[j]);
        float e   = eps[((size_t)b * NA_ + agent) * AD_ + j];
        float raw = mean[(size_t)b * HDP_ + j] + st * e;
        lp += -0.5f * e * e - logf(st);
        lp -= 2.f * (ln2 - raw - softplus_(-2.f * raw));
        float a = tanhf(raw);
        acts[((size_t)b * NA_ + agent) * AD_ + j] = a;
        raws[((size_t)b * NA_ + agent) * AD_ + j] = raw;
        aph[(size_t)b * APK_ + j] = (_Float16)a;   // cols [16..31] stay zero
    }
    logs[(size_t)b * NA_ + agent] = lp;
}

// ----------------------------- utility kernels -----------------------------
__global__ void zero_f32_kernel(float* p, size_t n) {
    size_t i = (size_t)blockIdx.x * blockDim.x + threadIdx.x;
    if (i < n) p[i] = 0.f;
}
__global__ void zero_f16_kernel(_Float16* p, size_t n) {
    size_t i = (size_t)blockIdx.x * blockDim.x + threadIdx.x;
    if (i < n) p[i] = (_Float16)0.f;
}
// Transposed f32->f16 weight convert: src [K,Nreal] row-major -> dst [Npad,Kpad]
// (Bt layout), zero-filling n in [Nreal,Npad) and k in [K,Kpad).
__global__ void cvtT_kernel(const float* __restrict__ s, _Float16* __restrict__ d,
                            int Nreal, int Npad, int K, int Kpad) {
    size_t i = (size_t)blockIdx.x * blockDim.x + threadIdx.x;
    if (i >= (size_t)Npad * Kpad) return;
    int n = (int)(i / Kpad), k = (int)(i % Kpad);
    d[i] = (n < Nreal && k < K) ? (_Float16)s[(size_t)k * Nreal + n] : (_Float16)0.f;
}
// rows = gridDim.y
__global__ void cvt16_strided_kernel(const float* s, int sld, _Float16* d, int dld, int cols) {
    int r = blockIdx.y;
    int c = blockIdx.x * blockDim.x + threadIdx.x;
    if (c < cols) d[(size_t)r * dld + c] = (_Float16)s[(size_t)r * sld + c];
}
__global__ void copy_rows_kernel(const float* s, int sld, float* d, int dld, int cols) {
    int r = blockIdx.y;
    int c = blockIdx.x * blockDim.x + threadIdx.x;
    if (c < cols) d[(size_t)r * dld + c] = s[(size_t)r * sld + c];
}

// ===========================================================================
extern "C" void kernel_launch(void* const* d_in, const int* in_sizes, int n_in,
                              void* d_out, int out_size, void* d_ws, size_t ws_size,
                              hipStream_t stream)
{
    (void)in_sizes; (void)n_in; (void)out_size; (void)ws_size;

    const float* obs_rep = (const float*)d_in[0];
    const float* obs     = (const float*)d_in[1];
    const float* eps     = (const float*)d_in[2];
    const float* W_embed = (const float*)d_in[3];
    const float* b_embed = (const float*)d_in[4];
    const float* W_obs   = (const float*)d_in[5];
    const float* ln1_s   = (const float*)d_in[6];
    const float* ln2_s   = (const float*)d_in[7];
    const float* ln_out  = (const float*)d_in[8];
    const float* W_head  = (const float*)d_in[9];
    const float* log_std = (const float*)d_in[10];
    const float* in_proj[2]  = {(const float*)d_in[11], (const float*)d_in[20]};
    const float* conv_w[2]   = {(const float*)d_in[12], (const float*)d_in[21]};
    const float* conv_b[2]   = {(const float*)d_in[13], (const float*)d_in[22]};
    const float* x_proj[2]   = {(const float*)d_in[14], (const float*)d_in[23]};
    const float* dt_proj[2]  = {(const float*)d_in[15], (const float*)d_in[24]};
    const float* dt_bias[2]  = {(const float*)d_in[16], (const float*)d_in[25]};
    const float* A_log[2]    = {(const float*)d_in[17], (const float*)d_in[26]};
    const float* Dvec[2]     = {(const float*)d_in[18], (const float*)d_in[27]};
    const float* out_proj[2] = {(const float*)d_in[19], (const float*)d_in[28]};

    float* acts = (float*)d_out;
    float* logs = acts + (size_t)B_ * NA_ * AD_;
    float* raws = logs + (size_t)B_ * NA_;

    // ---- workspace carve-up (256B aligned) ----
    char* cur = (char*)d_ws;
    auto alloc = [&](size_t bytes) -> void* {
        void* p = (void*)cur;
        cur += (bytes + 255) & ~(size_t)255;
        return p;
    };
    // f16 TRANSPOSED weight mirrors [Npad][Kpad] (L2-resident working set)
    _Float16* WembT  = (_Float16*)alloc((size_t)DM_ * APK_ * 2);
    _Float16* WobsT  = (_Float16*)alloc((size_t)DM_ * OD_ * 2);
    _Float16* WheadT = (_Float16*)alloc((size_t)HDP_ * DM_ * 2);
    _Float16 *inT[2], *xpT[2], *dtT[2], *outT[2];
    for (int w = 0; w < 2; ++w) {
        inT[w]  = (_Float16*)alloc((size_t)NB_ * 2 * DI_ * DM_ * 2);
        xpT[w]  = (_Float16*)alloc((size_t)NB_ * DBCP_ * DI_ * 2);
        dtT[w]  = (_Float16*)alloc((size_t)NB_ * DI_ * DR_ * 2);
        outT[w] = (_Float16*)alloc((size_t)NB_ * DM_ * DI_ * 2);
    }
    // recurrent f32 state
    float* hbuf[2], *cbuf[2];
    for (int w = 0; w < 2; ++w) {
        hbuf[w] = (float*)alloc((size_t)NB_ * B_ * DI_ * DS_ * 4);
        cbuf[w] = (float*)alloc((size_t)NB_ * B_ * DC_ * DI_ * 4);
    }
    // f32 activations
    float* xf     = (float*)alloc((size_t)B_ * DM_ * 4);
    float* ctxf   = (float*)alloc((size_t)B_ * DM_ * 4);
    float* xzf    = (float*)alloc((size_t)B_ * 2 * DI_ * 4);
    float* xcf    = (float*)alloc((size_t)B_ * DI_ * 4);
    float* dbcf   = (float*)alloc((size_t)B_ * DBCP_ * 4);
    float* deltaf = (float*)alloc((size_t)B_ * DI_ * 4);
    float* meanf  = (float*)alloc((size_t)B_ * HDP_ * 4);
    // f16 activations
    _Float16* u_h   = (_Float16*)alloc((size_t)B_ * DM_ * 2);
    _Float16* xc_h  = (_Float16*)alloc((size_t)B_ * DI_ * 2);
    _Float16* dt_h  = (_Float16*)alloc((size_t)B_ * DR_ * 2);
    _Float16* yg_h  = (_Float16*)alloc((size_t)B_ * DI_ * 2);
    _Float16* ap_h  = (_Float16*)alloc((size_t)B_ * APK_ * 2);
    _Float16* obs_h = (_Float16*)alloc((size_t)B_ * OD_ * 2);

    // ---- helpers ----
    auto gemm = [&](const _Float16* A, int lda, const _Float16* Bt, int ldbt,
                    float* C, int ldc, const float* bias,
                    int M, int N64, int K, int beta, int act) {
        int tiles  = (M / 16) * (N64 / 64);
        int blocks = (tiles + 7) / 8;  // 8 waves per 256-thread block
        wmma_gemm_kernel<<<blocks, 256, 0, stream>>>(A, lda, Bt, ldbt, C, ldc, bias,
                                                     M, N64, K, beta, act);
    };
    auto cvtT = [&](const float* s, _Float16* d, int Nr, int Np, int K, int Kp) {
        size_t n = (size_t)Np * Kp;
        cvtT_kernel<<<(unsigned)((n + 255) / 256), 256, 0, stream>>>(s, d, Nr, Np, K, Kp);
    };

    // ---- per-call init: transposed/padded f16 weights + zeroed state ----
    cvtT(W_embed, WembT, DM_, DM_, AD_, APK_);      // [512][32], K padded 16->32
    cvtT(W_obs,   WobsT, DM_, DM_, OD_, OD_);       // [512][128]
    cvtT(W_head,  WheadT, AD_, HDP_, DM_, DM_);     // [64][512], N padded 16->64
    for (int w = 0; w < 2; ++w) {
        for (int b = 0; b < NB_; ++b) {
            cvtT(in_proj[w]  + (size_t)b * DM_ * 2 * DI_, inT[w]  + (size_t)b * 2 * DI_ * DM_,
                 2 * DI_, 2 * DI_, DM_, DM_);
            cvtT(x_proj[w]   + (size_t)b * DI_ * 96,      xpT[w]  + (size_t)b * DBCP_ * DI_,
                 96, DBCP_, DI_, DI_);                    // N padded 96->128
            cvtT(dt_proj[w]  + (size_t)b * DR_ * DI_,     dtT[w]  + (size_t)b * DI_ * DR_,
                 DI_, DI_, DR_, DR_);
            cvtT(out_proj[w] + (size_t)b * DI_ * DM_,     outT[w] + (size_t)b * DM_ * DI_,
                 DM_, DM_, DI_, DI_);
        }
        size_t nh = (size_t)NB_ * B_ * DI_ * DS_;
        zero_f32_kernel<<<(unsigned)((nh + 255) / 256), 256, 0, stream>>>(hbuf[w], nh);
        size_t nb = (size_t)NB_ * B_ * DC_ * DI_;
        zero_f32_kernel<<<(unsigned)((nb + 255) / 256), 256, 0, stream>>>(cbuf[w], nb);
    }
    zero_f16_kernel<<<(B_ * APK_ + 255) / 256, 256, 0, stream>>>(ap_h, (size_t)B_ * APK_);

    // ---- one fused mamba step (w: 0 = self stream, 1 = ctx stream) ----
    auto mamba_step = [&](int w, int blk, const float* ln_scale, bool add_ctx) {
        ln_kernel<<<B_, 256, 0, stream>>>(xf, ln_scale, add_ctx ? ctxf : nullptr, u_h);
        gemm(u_h, DM_, inT[w] + (size_t)blk * 2 * DI_ * DM_, DM_,
             xzf, 2 * DI_, nullptr, B_, 2 * DI_, DM_, 0, 0);
        conv_kernel<<<(B_ * DI_) / 256, 256, 0, stream>>>(
            cbuf[w] + (size_t)blk * B_ * DC_ * DI_, xzf,
            conv_w[w] + (size_t)blk * DC_ * DI_, conv_b[w] + (size_t)blk * DI_, xcf, xc_h);
        gemm(xc_h, DI_, xpT[w] + (size_t)blk * DBCP_ * DI_, DI_,
             dbcf, DBCP_, nullptr, B_, DBCP_, DI_, 0, 0);
        cvt16_strided_kernel<<<dim3(1, B_), 256, 0, stream>>>(dbcf, DBCP_, dt_h, DR_, DR_);
        gemm(dt_h, DR_, dtT[w] + (size_t)blk * DI_ * DR_, DR_,
             deltaf, DI_, dt_bias[w] + (size_t)blk * DI_, B_, DI_, DR_, 0, /*softplus*/1);
        state_kernel<<<(B_ * DI_) / 256, 256, 0, stream>>>(
            hbuf[w] + (size_t)blk * B_ * DI_ * DS_,
            A_log[w] + (size_t)blk * DI_ * DS_, Dvec[w] + (size_t)blk * DI_,
            dbcf, deltaf, xcf, xzf, yg_h);
        gemm(yg_h, DI_, outT[w] + (size_t)blk * DM_ * DI_, DI_,
             xf, DM_, nullptr, B_, DM_, DI_, /*beta: x += o*/1, 0);
    };

    // ---- sequential agent loop ----
    for (int i = 0; i < NA_; ++i) {
        // ctx = obs_rep[:,i] + obs[:,i] @ W_obs
        cvt16_strided_kernel<<<dim3(1, B_), 256, 0, stream>>>(
            obs + (size_t)i * OD_, NA_ * OD_, obs_h, OD_, OD_);
        copy_rows_kernel<<<dim3(2, B_), 256, 0, stream>>>(
            obs_rep + (size_t)i * DM_, NA_ * DM_, ctxf, DM_, DM_);
        gemm(obs_h, OD_, WobsT, OD_, ctxf, DM_, nullptr, B_, DM_, OD_, 1, 0);
        // x = a_prev @ W_embed + b_embed   (K padded to 32)
        gemm(ap_h, APK_, WembT, APK_, xf, DM_, b_embed, B_, DM_, APK_, 0, 0);

        for (int b = 0; b < NB_; ++b) {
            mamba_step(0, b, ln1_s + (size_t)b * DM_, false);
            mamba_step(1, b, ln2_s + (size_t)b * DM_, true);
        }

        // head
        ln_kernel<<<B_, 256, 0, stream>>>(xf, ln_out, nullptr, u_h);
        gemm(u_h, DM_, WheadT, DM_, meanf, HDP_, nullptr, B_, HDP_, DM_, 0, 0);
        head_kernel<<<1, 128, 0, stream>>>(meanf, eps, log_std, acts, logs, raws, ap_h, i);
    }
}
